// FusionFeedForward_12764642803785
// MI455X (gfx1250) — matrix operand order repaired and checked by reading, not compiled, not run on hardware
//
#include <hip/hip_runtime.h>
#include <hip/hip_bf16.h>

// ---------------- problem constants (from reference) ----------------
#define MROWS 11520   // b*n = 2*5760
#define DDIM  512     // d
#define HDIM  1960    // hd
#define HDP   1984    // hd padded to multiple of 32 (WMMA K) and 16 (N tiles)
#define B2N   16      // b*n / NVECS
#define CCH   40      // hd / 49
#define HH    60
#define WWW   108
#define LHN   20
#define LWN   36
#define NVECS 720     // LHN*LWN

typedef __bf16 bf16_t;
typedef __attribute__((ext_vector_type(16))) __bf16 v16bf;
typedef __attribute__((ext_vector_type(8)))  __bf16 v8bf;
typedef __attribute__((ext_vector_type(8)))  float  v8f;

// ---------------- conversion kernels ----------------
__global__ __launch_bounds__(256) void cvt_x_kernel(const float* __restrict__ x,
                                                    bf16_t* __restrict__ xb) {
    size_t i = (size_t)blockIdx.x * 256 + threadIdx.x;
    if (i < (size_t)MROWS * DDIM) xb[i] = (bf16_t)x[i];
}

// w1t[n][k] = n < HDIM ? W1[k][n] : 0     (W1 is DDIM x HDIM row-major)
__global__ __launch_bounds__(256) void cvt_w1t_kernel(const float* __restrict__ W1,
                                                      bf16_t* __restrict__ w1t) {
    size_t i = (size_t)blockIdx.x * 256 + threadIdx.x;
    if (i >= (size_t)HDP * DDIM) return;
    int k = (int)(i % DDIM);
    int n = (int)(i / DDIM);
    w1t[i] = (n < HDIM) ? (bf16_t)W1[(size_t)k * HDIM + n] : (bf16_t)0.0f;
}

// w2t[n][k] = k < HDIM ? W2[k][n] : 0     (W2 is HDIM x DDIM row-major)
__global__ __launch_bounds__(256) void cvt_w2t_kernel(const float* __restrict__ W2,
                                                      bf16_t* __restrict__ w2t) {
    size_t i = (size_t)blockIdx.x * 256 + threadIdx.x;
    if (i >= (size_t)DDIM * HDP) return;
    int k = (int)(i % HDP);
    int n = (int)(i / HDP);
    w2t[i] = (k < HDIM) ? (bf16_t)W2[(size_t)k * DDIM + n] : (bf16_t)0.0f;
}

// ---------------- CDNA5 async global->LDS copy (ASYNCcnt) ----------------
// LDS aperture rule (ISA 10.2): the LDS address is the low 32 bits of the flat address, so truncating the
// generic pointer of a __shared__ element yields the LDS byte offset for VDST.
__device__ __forceinline__ void async_copy_b128(bf16_t* lds, const bf16_t* g) {
    unsigned int l = (unsigned int)(size_t)lds;
    asm volatile("global_load_async_to_lds_b128 %0, %1, off"
                 :: "v"(l), "v"(g) : "memory");
}
__device__ __forceinline__ void wait_async0() {
    asm volatile("s_wait_asynccnt 0x0" ::: "memory");
}

// Stage one 128x32 A tile + 64x32 B tile into an LDS buffer (3 async b128/thread)
__device__ __forceinline__ void stage_async(const bf16_t* __restrict__ A,
                                            const bf16_t* __restrict__ B,
                                            bf16_t* __restrict__ AsBuf,
                                            bf16_t* __restrict__ BsBuf,
                                            int tid, int m0, int n0, int K, int k0) {
    int e = tid * 8;                           // 8 bf16 = 16 B per op
    int r = e >> 5, c = e & 31;                // A elems [0,2048)
    async_copy_b128(&AsBuf[r * 56 + c], &A[(size_t)(m0 + r) * K + k0 + c]);
    int e2 = e + 2048;                         // A elems [2048,4096)
    int r2 = e2 >> 5, c2 = e2 & 31;
    async_copy_b128(&AsBuf[r2 * 56 + c2], &A[(size_t)(m0 + r2) * K + k0 + c2]);
    async_copy_b128(&BsBuf[r * 56 + c], &B[(size_t)(n0 + r) * K + k0 + c]); // r<64
}

// ---------------- WMMA GEMM: C = A * B^T + bias ----------------
// A: M x K row-major bf16 (lda = K), B: N x K row-major bf16 (ldb = K)
// Block tile 128(M) x 64(N), 8 waves as 4(M) x 2(N), wave tile 32x32.
// Double-buffered LDS, async global->LDS, one barrier per 32-K step.
template <bool OUT_BF16>
__global__ __launch_bounds__(256) void gemm_wmma_kernel(
    const bf16_t* __restrict__ A, const bf16_t* __restrict__ B,
    const float* __restrict__ bias, int biasLen,
    void* __restrict__ Cout, int M, int N, int K, int ldc) {

    __shared__ bf16_t As[2][128 * 56];   // stride 56 elems = 112 B (16B aligned)
    __shared__ bf16_t Bs[2][64 * 56];

    const int tid   = threadIdx.x;
    const int lane  = tid & 31;
    const int wid   = tid >> 5;
    const int waveM = wid & 3;        // 0..3
    const int waveN = wid >> 2;       // 0..1
    const int m0 = blockIdx.y * 128;
    const int n0 = blockIdx.x * 64;

    v8f acc[2][2];
    const v8f vzero = {0.f, 0.f, 0.f, 0.f, 0.f, 0.f, 0.f, 0.f};
    acc[0][0] = vzero; acc[0][1] = vzero; acc[1][0] = vzero; acc[1][1] = vzero;

    const int aRow  = lane & 15;
    const int half  = lane >> 4;      // 0 for lanes 0-15, 1 for lanes 16-31
    const int koffA = half * 8;       // A frag: K runs {0-7,16-23} / {8-15,24-31}
    const int koffB = half * 8;       // B frag: K runs {0-7,16-23} / {8-15,24-31}

    const int nIter = K >> 5;
    stage_async(A, B, As[0], Bs[0], tid, m0, n0, K, 0);

    for (int it = 0; it < nIter; ++it) {
        const int buf = it & 1;
        wait_async0();         // my stage(it) writes landed in LDS
        __syncthreads();       // everyone's writes visible; buf^1 reads retired

        if (it + 1 < nIter)    // uniform predicate (EXEC stays all-ones)
            stage_async(A, B, As[buf ^ 1], Bs[buf ^ 1], tid, m0, n0, K,
                        (it + 1) * 32);

        // ---- build operand fragments from LDS ----
        v16bf afrag[2], bfrag[2];
        #pragma unroll
        for (int ti = 0; ti < 2; ++ti) {
            union { v16bf v; v8bf h[2]; } u;
            const bf16_t* p = &As[buf][(waveM * 32 + ti * 16 + aRow) * 56];
            u.h[0] = *(const v8bf*)(p + koffA);
            u.h[1] = *(const v8bf*)(p + 16 + koffA);
            afrag[ti] = u.v;
        }
        #pragma unroll
        for (int tj = 0; tj < 2; ++tj) {
            union { v16bf v; v8bf h[2]; } u;
            const bf16_t* p = &Bs[buf][(waveN * 32 + tj * 16 + aRow) * 56 + koffB];
            u.h[0] = *(const v8bf*)(p);
            u.h[1] = *(const v8bf*)(p + 16);
            bfrag[tj] = u.v;
        }

        // ---- matrix core: 4x v_wmma_f32_16x16x32_bf16 ----
        #pragma unroll
        for (int ti = 0; ti < 2; ++ti)
            #pragma unroll
            for (int tj = 0; tj < 2; ++tj)
                acc[ti][tj] = __builtin_amdgcn_wmma_f32_16x16x32_bf16(
                    false, afrag[ti], false, bfrag[tj],
                    (short)0, acc[ti][tj], false, false);
    }

    // ---- epilogue: bias add + store (C layout: VGPR r -> M=r+8*half, N=lane&15)
    const int rOff = half * 8;
    const int colL = lane & 15;
    #pragma unroll
    for (int ti = 0; ti < 2; ++ti) {
        #pragma unroll
        for (int tj = 0; tj < 2; ++tj) {
            int col = n0 + waveN * 32 + tj * 16 + colL;
            float bv = (col < biasLen) ? bias[col] : 0.0f;
            #pragma unroll
            for (int r = 0; r < 8; ++r) {
                int row = m0 + waveM * 32 + ti * 16 + rOff + r;
                float v = acc[ti][tj][r] + bv;
                if (OUT_BF16)
                    ((bf16_t*)Cout)[(size_t)row * ldc + col] = (bf16_t)v;
                else
                    ((float*)Cout)[(size_t)row * ldc + col] = v;
            }
        }
    }
}

// ---------------- fold: y[b2,cc,i,j] = sum(patches)/count ----------------
__global__ __launch_bounds__(256) void fold_kernel(const bf16_t* __restrict__ hb,
                                                   float* __restrict__ y) {
    int idx = blockIdx.x * 256 + threadIdx.x;
    if (idx >= B2N * CCH * HH * WWW) return;
    int j = idx % WWW; int t = idx / WWW;
    int i = t % HH;    t /= HH;
    int cc = t % CCH;  int b2 = t / CCH;
    int ip = i + 3, jp = j + 3;

    int cntR = 0, cntC = 0;
    for (int ky = ip % 3; ky < 7; ky += 3) { int lh = (ip - ky) / 3; if (lh < LHN) cntR++; }
    for (int kx = jp % 3; kx < 7; kx += 3) { int lw = (jp - kx) / 3; if (lw < LWN) cntC++; }

    float sum = 0.0f;
    for (int ky = ip % 3; ky < 7; ky += 3) {
        int lh = (ip - ky) / 3;
        if (lh >= LHN) continue;
        for (int kx = jp % 3; kx < 7; kx += 3) {
            int lw = (jp - kx) / 3;
            if (lw >= LWN) continue;
            int row = b2 * NVECS + lh * LWN + lw;
            int col = cc * 49 + ky * 7 + kx;
            sum += (float)hb[(size_t)row * HDP + col];
        }
    }
    y[idx] = sum / (float)(cntR * cntC);
}

// ---------------- unfold + ReLU -> bf16 (zero-padded to HDP cols) ----------------
__global__ __launch_bounds__(256) void unfold_kernel(const float* __restrict__ y,
                                                     bf16_t* __restrict__ uf) {
    size_t idx = (size_t)blockIdx.x * 256 + threadIdx.x;
    if (idx >= (size_t)MROWS * HDP) return;
    int col = (int)(idx % HDP);
    int row = (int)(idx / HDP);
    float v = 0.0f;
    if (col < HDIM) {
        int cc = col / 49, rem = col % 49, ky = rem / 7, kx = rem % 7;
        int b2 = row / NVECS, vv = row % NVECS, lh = vv / LWN, lw = vv % LWN;
        int r = ky + 3 * lh - 3, c = kx + 3 * lw - 3;
        if (r >= 0 && r < HH && c >= 0 && c < WWW)
            v = y[((size_t)(b2 * CCH + cc) * HH + r) * WWW + c];
    }
    uf[idx] = (bf16_t)fmaxf(v, 0.0f);
}

// ---------------- launch ----------------
extern "C" void kernel_launch(void* const* d_in, const int* in_sizes, int n_in,
                              void* d_out, int out_size, void* d_ws, size_t ws_size,
                              hipStream_t stream) {
    const float* x  = (const float*)d_in[0];
    const float* W1 = (const float*)d_in[1];
    const float* b1 = (const float*)d_in[2];
    const float* W2 = (const float*)d_in[3];
    const float* b2 = (const float*)d_in[4];

    size_t off = 0;
    auto ws_alloc = [&](size_t bytes) {
        void* p = (char*)d_ws + off;
        off += (bytes + 255) & ~(size_t)255;
        return p;
    };
    bf16_t* xb  = (bf16_t*)ws_alloc((size_t)MROWS * DDIM * sizeof(bf16_t));
    bf16_t* w1t = (bf16_t*)ws_alloc((size_t)HDP * DDIM * sizeof(bf16_t));
    bf16_t* hb  = (bf16_t*)ws_alloc((size_t)MROWS * HDP * sizeof(bf16_t));
    float*  yb  = (float*) ws_alloc((size_t)B2N * CCH * HH * WWW * sizeof(float));
    bf16_t* uf  = (bf16_t*)ws_alloc((size_t)MROWS * HDP * sizeof(bf16_t));
    bf16_t* w2t = (bf16_t*)ws_alloc((size_t)DDIM * HDP * sizeof(bf16_t));
    (void)ws_size; (void)in_sizes; (void)n_in; (void)out_size;

    // bf16 conversions (+ weight transposes for [N][K] operand layout)
    cvt_x_kernel  <<<(MROWS * DDIM) / 256, 256, 0, stream>>>(x, xb);
    cvt_w1t_kernel<<<(HDP * DDIM) / 256, 256, 0, stream>>>(W1, w1t);
    cvt_w2t_kernel<<<(DDIM * HDP) / 256, 256, 0, stream>>>(W2, w2t);

    // GEMM1: h = x @ W1 + b1  -> bf16, row stride HDP (pad cols are zero)
    gemm_wmma_kernel<true><<<dim3(HDP / 64, MROWS / 128), 256, 0, stream>>>(
        xb, w1t, b1, HDIM, hb, MROWS, HDP, DDIM, HDP);

    // fold + normalize
    fold_kernel<<<(B2N * CCH * HH * WWW + 255) / 256, 256, 0, stream>>>(hb, yb);

    // unfold + ReLU -> bf16 (zero pad cols [HDIM,HDP))
    unfold_kernel<<<(size_t)(MROWS * (size_t)HDP) / 256, 256, 0, stream>>>(yb, uf);

    // GEMM2: out = relu(uf) @ W2 + b2 -> fp32 d_out
    gemm_wmma_kernel<false><<<dim3(DDIM / 64, MROWS / 128), 256, 0, stream>>>(
        uf, w2t, b2, DDIM, d_out, MROWS, DDIM, HDP, DDIM);
}
